// SambaMixerBlock_16784732193346
// MI455X (gfx1250) — compile-verified, matrix operand order checked by reading, not run on hardware
//
#include <hip/hip_runtime.h>
#include <cstdint>

// ---------------- problem dims ----------------
#define BSZ   8
#define SEQ   1024
#define DM    1024
#define NST   16
#define NCONV 4
#define DI    2048          // EXPAND * D
#define DTR   64            // dt_rank
#define GCOLS 96            // DTR + 2*NST
#define MROWS (BSZ * SEQ)   // 8192 rows for every GEMM
#define EPSV  1e-5f
#define KCHUNK 256          // K-chunk staged into LDS by the TDM

typedef unsigned int u32;
typedef __attribute__((ext_vector_type(16))) __bf16 v16bf;
typedef __attribute__((ext_vector_type(8)))  float  v8f;
typedef __attribute__((ext_vector_type(4)))  u32    u32x4;
typedef __attribute__((ext_vector_type(8)))  int    i32x8;
typedef __attribute__((ext_vector_type(4)))  int    i32x4;

// ---------------- small helpers ----------------
__global__ __launch_bounds__(256) void cvt_f32_bf16_kernel(const float* __restrict__ s,
                                                           __bf16* __restrict__ d, size_t n) {
  size_t i = (size_t)blockIdx.x * blockDim.x + threadIdx.x;
  if (i < n) d[i] = (__bf16)s[i];
}

// set bits [lo, lo+nbits) of a little-endian u32 word array (pre-zeroed)
__device__ inline void desc_set(u32* w, int lo, int nbits, unsigned long long v) {
  for (int i = 0; i < nbits; ++i) {
    int bit = lo + i;
    if ((v >> i) & 1ull) w[bit >> 5] |= (1u << (bit & 31));
  }
}

// issue a TDM load of a (rows x kc) bf16 tile of W (row-major, stride ldw elements)
// starting at element (tn, k0), into LDS offset 0 (dynamic-LDS base).
__device__ inline void tdm_load_w_tile(const __bf16* W, int tn, int k0, int rows,
                                       int kc, int Nc, int K, int ldw)
{
#if defined(__has_builtin) && __has_builtin(__builtin_amdgcn_tensor_load_to_lds)
  u32 g1w[8] = {0, 0, 0, 0, 0, 0, 0, 0};
  desc_set(g1w, 16, 2, 1ull);                           // data_size = 2 bytes
  desc_set(g1w, 48, 32, (unsigned long long)(u32)K);    // tensor_dim0
  desc_set(g1w, 80, 32, (unsigned long long)(u32)Nc);   // tensor_dim1
  desc_set(g1w, 112, 16, (unsigned long long)(u32)kc);  // tile_dim0
  desc_set(g1w, 128, 16, (unsigned long long)(u32)rows);// tile_dim1
  desc_set(g1w, 160, 48, (unsigned long long)(u32)ldw); // tensor_dim0_stride
  unsigned long long gaddr =
      (unsigned long long)(uintptr_t)(W + (size_t)tn * ldw + k0);
  u32x4 g0;
  g0.x = 1u;                                            // count = 1 descriptor
  g0.y = 0u;                                            // lds_addr = dynamic-LDS base
  g0.z = (u32)(gaddr & 0xffffffffu);
  g0.w = (u32)((gaddr >> 32) & 0x01ffffffu) | (2u << 30);  // type=2 ("image")
  i32x8 g1;
#pragma unroll
  for (int i = 0; i < 8; ++i) g1[i] = (int)g1w[i];
  i32x4 zz4 = {0, 0, 0, 0};
  i32x8 zz8 = {0, 0, 0, 0, 0, 0, 0, 0};
  __builtin_amdgcn_tensor_load_to_lds(g0, g1, zz4, zz4, zz8, 0);
  __builtin_amdgcn_s_wait_tensorcnt(0);
#else
  (void)W; (void)tn; (void)k0; (void)rows; (void)kc; (void)Nc; (void)K; (void)ldw;
#endif
}

// ---------------- WMMA GEMM: C[M,Nc] f32 = A[M,K](bf16) x W[Nc,K]^T(bf16) ----------------
// plain version: each wave owns one 16x16 tile, operands streamed from global
// (used for the small-N x-proj and the K=64 dt-proj where L2 holds W trivially)
__global__ __launch_bounds__(256)
void wmma_gemm_bf16(const __bf16* __restrict__ A, const __bf16* __restrict__ W,
                    float* __restrict__ C, int M, int Nc, int K,
                    int lda, int ldw, int ldc)
{
  int wave = (blockIdx.x * blockDim.x + threadIdx.x) >> 5;
  int lane = threadIdx.x & 31;
  int ntn  = Nc >> 4;
  int tm   = (wave / ntn) << 4;
  int tn   = (wave % ntn) << 4;
  if (tm >= M) return;

  const __bf16* Ap = A + (size_t)(tm + (lane & 15)) * lda + ((lane >> 4) << 4);
  const __bf16* Wp = W + (size_t)(tn + (lane & 15)) * ldw + ((lane >> 4) << 4);
  v8f acc = {0.f, 0.f, 0.f, 0.f, 0.f, 0.f, 0.f, 0.f};

  for (int k = 0; k < K; k += 64) {
    __builtin_prefetch((const void*)(Ap + k + 256), 0, 0);   // global_prefetch
    v16bf a0 = *(const v16bf*)(Ap + k);
    v16bf b0 = *(const v16bf*)(Wp + k);
    acc = __builtin_amdgcn_wmma_f32_16x16x32_bf16(false, a0, false, b0, (short)0, acc, false, false);
    v16bf a1 = *(const v16bf*)(Ap + k + 32);
    v16bf b1 = *(const v16bf*)(Wp + k + 32);
    acc = __builtin_amdgcn_wmma_f32_16x16x32_bf16(false, a1, false, b1, (short)0, acc, false, false);
  }

  int col  = tn + (lane & 15);
  int row0 = tm + ((lane >> 4) << 3);
#pragma unroll
  for (int j = 0; j < 8; ++j)
    C[(size_t)(row0 + j) * ldc + col] = acc[j];
}

// TDM version: 128(M) x 64(N) block tile, 8 waves. Each wave keeps 4 f32
// accumulators (16M x 64N), reuses one A fragment across 4 WMMAs; the 64 x
// KCHUNK weight tile lives in LDS, staged chunk-by-chunk by the Tensor Data
// Mover (wave 0 issues tensor_load_to_lds + s_wait_tensorcnt, barrier shares).
__global__ __launch_bounds__(256)
void wmma_gemm_bf16_tdm(const __bf16* __restrict__ A, const __bf16* __restrict__ W,
                        float* __restrict__ C, int M, int Nc, int K,
                        int lda, int ldw, int ldc)
{
  extern __shared__ __bf16 sW[];            // 64 x KCHUNK bf16 tile (row-major)
  int wv   = threadIdx.x >> 5;
  int lane = threadIdx.x & 31;
  int tn   = blockIdx.x << 6;               // 64 N-columns per block
  int tm   = ((blockIdx.y << 3) + wv) << 4; // 16 M-rows per wave
  (void)M;

  const __bf16* Ap = A + (size_t)(tm + (lane & 15)) * lda + ((lane >> 4) << 4);
  v8f acc[4];
#pragma unroll
  for (int j = 0; j < 4; ++j)
    acc[j] = (v8f){0.f, 0.f, 0.f, 0.f, 0.f, 0.f, 0.f, 0.f};

  for (int kc = 0; kc < K; kc += KCHUNK) {
#if defined(__has_builtin) && __has_builtin(__builtin_amdgcn_tensor_load_to_lds)
    if (wv == 0)
      tdm_load_w_tile(W, tn, kc, 64, KCHUNK, Nc, K, ldw);
#else
    for (int i = threadIdx.x; i < 64 * KCHUNK; i += blockDim.x)
      sW[i] = W[(size_t)(tn + (i / KCHUNK)) * ldw + kc + (i % KCHUNK)];
#endif
    __syncthreads();

    __builtin_prefetch((const void*)(Ap + kc + KCHUNK), 0, 0);  // next A chunk
#pragma unroll 2
    for (int k = 0; k < KCHUNK; k += 32) {
      v16bf a = *(const v16bf*)(Ap + kc + k);
#pragma unroll
      for (int j = 0; j < 4; ++j) {
        const __bf16* Wp =
            sW + (size_t)((j << 4) + (lane & 15)) * KCHUNK + ((lane >> 4) << 4) + k;
        v16bf bb = *(const v16bf*)Wp;                    // ds_load_b128 x2
        acc[j] = __builtin_amdgcn_wmma_f32_16x16x32_bf16(false, a, false, bb,
                                                         (short)0, acc[j], false, false);
      }
    }
    __syncthreads();
  }

  int row0 = tm + ((lane >> 4) << 3);
#pragma unroll
  for (int j = 0; j < 4; ++j) {
    int col = tn + (j << 4) + (lane & 15);
#pragma unroll
    for (int r = 0; r < 8; ++r)
      C[(size_t)(row0 + r) * ldc + col] = acc[j][r];
  }
}

// ---------------- prep token-mixer: res1, RMS, softmax-weighted mix ----------------
__global__ __launch_bounds__(256)
void prep_tm_kernel(const float* __restrict__ hstm, const float* __restrict__ hscm,
                    const float* __restrict__ residual,
                    const float* __restrict__ norm_w, const float* __restrict__ wavg,
                    float* __restrict__ res1, __bf16* __restrict__ xmix)
{
  int bs = blockIdx.x;                 // b*SEQ + s
  int b = bs / SEQ, s = bs % SEQ;
  const size_t slc = (size_t)SEQ * DM;
  const float* h0 = hstm + ((size_t)b * 2) * slc + (size_t)s * DM;
  const float* h1 = h0 + slc;
  const float* c0 = hscm + ((size_t)b * 2) * slc + (size_t)s * DM;
  const float* c1 = c0 + slc;
  const float* rv = residual + (size_t)bs * DM;
  float* r1 = res1 + (size_t)bs * DM;

  __shared__ float srow[DM];
  __shared__ float sred[8];
  __shared__ float snorm;

  float ss = 0.f;
  for (int d = threadIdx.x; d < DM; d += blockDim.x) {
    float r = rv[d] + c1[d];
    srow[d] = r; r1[d] = r; ss += r * r;
  }
  for (int o = 16; o > 0; o >>= 1) ss += __shfl_down(ss, o, 32);
  if ((threadIdx.x & 31) == 0) sred[threadIdx.x >> 5] = ss;
  __syncthreads();
  if (threadIdx.x == 0) {
    float t = 0.f;
    for (int i = 0; i < 8; ++i) t += sred[i];
    snorm = rsqrtf(t / (float)DM + EPSV);
  }
  __syncthreads();
  float rms = snorm;

  float w0 = wavg[0 * SEQ + s], w1 = wavg[1 * SEQ + s];
  float w2 = wavg[2 * SEQ + s], w3 = wavg[3 * SEQ + s];
  float mx = fmaxf(fmaxf(w0, w1), fmaxf(w2, w3));
  float e0 = __expf(w0 - mx), e1 = __expf(w1 - mx), e2 = __expf(w2 - mx), e3 = __expf(w3 - mx);
  float inv = 1.f / (e0 + e1 + e2 + e3);
  w0 = e0 * inv; w1 = e1 * inv; w2 = e2 * inv; w3 = e3 * inv;

  for (int d = threadIdx.x; d < DM; d += blockDim.x) {
    float hn = srow[d] * rms * norm_w[d];
    float x = w0 * h0[d] + w1 * h1[d] + w2 * c0[d] + w3 * hn;
    xmix[(size_t)bs * DM + d] = (__bf16)x;
  }
}

// ---------------- prep channel-mixer: res2(out), RMS, 5-way mix, transposed bf16 ----------------
__global__ __launch_bounds__(256)
void prep_cm_kernel(const float* __restrict__ hstm, const float* __restrict__ hscm,
                    const float* __restrict__ res1, const float* __restrict__ out_tm,
                    const float* __restrict__ norm_w, const float* __restrict__ wavg,
                    float* __restrict__ res2, __bf16* __restrict__ xcmT)
{
  int bs = blockIdx.x;
  int b = bs / SEQ, s = bs % SEQ;
  const size_t slc = (size_t)SEQ * DM;
  const float* h0 = hstm + ((size_t)b * 2) * slc + (size_t)s * DM;
  const float* h1 = h0 + slc;
  const float* c0 = hscm + ((size_t)b * 2) * slc + (size_t)s * DM;
  const float* c1 = c0 + slc;

  __shared__ float srow[DM];
  __shared__ float sred[8];
  __shared__ float snorm;

  float ss = 0.f;
  for (int d = threadIdx.x; d < DM; d += blockDim.x) {
    float r = res1[(size_t)bs * DM + d] + out_tm[(size_t)bs * DM + d];
    srow[d] = r; res2[(size_t)bs * DM + d] = r; ss += r * r;
  }
  for (int o = 16; o > 0; o >>= 1) ss += __shfl_down(ss, o, 32);
  if ((threadIdx.x & 31) == 0) sred[threadIdx.x >> 5] = ss;
  __syncthreads();
  if (threadIdx.x == 0) {
    float t = 0.f;
    for (int i = 0; i < 8; ++i) t += sred[i];
    snorm = rsqrtf(t / (float)DM + EPSV);
  }
  __syncthreads();
  float rms = snorm;

  float w[5];
  float mx = -1e30f;
  for (int n = 0; n < 5; ++n) { w[n] = wavg[n * SEQ + s]; mx = fmaxf(mx, w[n]); }
  float tot = 0.f;
  for (int n = 0; n < 5; ++n) { w[n] = __expf(w[n] - mx); tot += w[n]; }
  float inv = 1.f / tot;
  for (int n = 0; n < 5; ++n) w[n] *= inv;

  for (int d = threadIdx.x; d < DM; d += blockDim.x) {
    float hn = srow[d] * rms * norm_w[d];
    float x = w[0] * h0[d] + w[1] * h1[d] + w[2] * hn + w[3] * c0[d] + w[4] * c1[d];
    xcmT[((size_t)b * DM + d) * SEQ + s] = (__bf16)x;   // transposed: (B, D, S)
  }
}

// ---------------- depthwise causal conv (DCONV=4) + SiLU ----------------
__global__ __launch_bounds__(256)
void conv_silu_kernel(const float* __restrict__ xz, const float* __restrict__ cw,
                      const float* __restrict__ cb, float* __restrict__ xf,
                      __bf16* __restrict__ xb, int L, int ldxz)
{
  size_t i = (size_t)blockIdx.x * blockDim.x + threadIdx.x;   // over B*L*DI
  int d = (int)(i % DI);
  size_t bt = i / DI;
  int t = (int)(bt % L);
  size_t b = bt / L;
  float acc = cb[d];
#pragma unroll
  for (int j = 0; j < NCONV; ++j) {
    int tt = t - (NCONV - 1) + j;
    if (tt >= 0) acc += cw[d * NCONV + j] * xz[(b * L + tt) * (size_t)ldxz + d];
  }
  float sv = acc / (1.f + __expf(-acc));
  xf[i] = sv;
  xb[i] = (__bf16)sv;
}

// ---------------- dt = softplus(dt_pre + dt_b) ----------------
__global__ __launch_bounds__(256)
void softplus_kernel(float* __restrict__ dt, const float* __restrict__ dtb, size_t n)
{
  size_t i = (size_t)blockIdx.x * blockDim.x + threadIdx.x;
  if (i >= n) return;
  float x = dt[i] + dtb[i % DI];
  dt[i] = (x > 20.f) ? x : log1pf(__expf(x));
}

// ---------------- selective scan + D skip + SiLU(z) gate -> bf16 y ----------------
__global__ __launch_bounds__(256)
void scan_kernel(const float* __restrict__ dt, const float* __restrict__ xdbl,
                 const float* __restrict__ xh, const float* __restrict__ A_log,
                 const float* __restrict__ D_p, const float* __restrict__ xz, int ldxz,
                 __bf16* __restrict__ yb, int L)
{
  int d = blockIdx.x * blockDim.x + threadIdx.x;   // channel within DI
  int b = blockIdx.y;
  float A[NST];
#pragma unroll
  for (int n = 0; n < NST; ++n) A[n] = -__expf(A_log[(size_t)d * NST + n]);
  float Dv = D_p[d];
  float h[NST];
#pragma unroll
  for (int n = 0; n < NST; ++n) h[n] = 0.f;

  __shared__ float sBC[2 * NST];
  for (int t = 0; t < L; ++t) {
    size_t bt = (size_t)b * L + t;
    if (threadIdx.x < 2 * NST)
      sBC[threadIdx.x] = xdbl[bt * GCOLS + DTR + threadIdx.x];
    __syncthreads();
    float dtv = dt[bt * DI + d];
    float xv  = xh[bt * DI + d];
    float dx  = dtv * xv;
    float y = 0.f;
#pragma unroll
    for (int n = 0; n < NST; ++n) {
      h[n] = __expf(dtv * A[n]) * h[n] + dx * sBC[n];
      y += h[n] * sBC[NST + n];
    }
    float zv = xz[bt * (size_t)ldxz + DI + d];
    float yo = (y + xv * Dv) * (zv / (1.f + __expf(-zv)));
    yb[bt * DI + d] = (__bf16)yo;
    __syncthreads();
  }
}

// ---------------- final (B,D,S) -> (B,S,D) transpose into d_out ----------------
__global__ __launch_bounds__(256)
void transpose_out_kernel(const float* __restrict__ src, float* __restrict__ dst)
{
  size_t i = (size_t)blockIdx.x * blockDim.x + threadIdx.x;   // over B*S*D (dst index)
  int d = (int)(i % DM);
  size_t t = i / DM;
  int s = (int)(t % SEQ);
  int b = (int)(t / SEQ);
  dst[i] = src[((size_t)b * DM + d) * SEQ + s];
}

// ---------------- host-side: one full Mamba block ----------------
static void run_mamba(hipStream_t stream, const __bf16* x_bf,
                      const float* in_w, const float* conv_w, const float* conv_b,
                      const float* xproj_w, const float* dt_w, const float* dt_b,
                      const float* A_log, const float* D_p, const float* out_w,
                      __bf16* wbuf, float* xz, float* xhf, __bf16* xhb,
                      float* xdblf, __bf16* xdblb, float* dt, __bf16* yb, float* outp)
{
  const int M = MROWS;
  const size_t ldsTile = 64 * KCHUNK * sizeof(__bf16);   // 32 KB staged weight tile
  // in-projection: xz = x @ in_w^T   (M x 4096, K=1024)
  cvt_f32_bf16_kernel<<<(2 * DI * DM) / 256, 256, 0, stream>>>(in_w, wbuf, (size_t)2 * DI * DM);
  {
    dim3 g((2 * DI) / 64, M / 128);
    wmma_gemm_bf16_tdm<<<g, 256, ldsTile, stream>>>(
        x_bf, wbuf, xz, M, 2 * DI, DM, DM, DM, 2 * DI);
  }
  // depthwise conv + SiLU on xh half
  conv_silu_kernel<<<(M * DI) / 256, 256, 0, stream>>>(xz, conv_w, conv_b, xhf, xhb, SEQ, 2 * DI);
  // x_dbl = xh @ xproj^T   (M x 96, K=2048)
  cvt_f32_bf16_kernel<<<(GCOLS * DI) / 256, 256, 0, stream>>>(xproj_w, wbuf, (size_t)GCOLS * DI);
  {
    int waves = (M / 16) * (GCOLS / 16);
    wmma_gemm_bf16<<<waves * 32 / 256, 256, 0, stream>>>(xhb, wbuf, xdblf, M, GCOLS, DI, DI, DI, GCOLS);
  }
  cvt_f32_bf16_kernel<<<(M * GCOLS) / 256, 256, 0, stream>>>(xdblf, xdblb, (size_t)M * GCOLS);
  // dt = softplus(x_dbl[:, :64] @ dt_w^T + dt_b)   (M x 2048, K=64)
  cvt_f32_bf16_kernel<<<(DI * DTR) / 256, 256, 0, stream>>>(dt_w, wbuf, (size_t)DI * DTR);
  {
    int waves = (M / 16) * (DI / 16);
    wmma_gemm_bf16<<<waves * 32 / 256, 256, 0, stream>>>(xdblb, wbuf, dt, M, DI, DTR, GCOLS, DTR, DI);
  }
  softplus_kernel<<<(M * DI) / 256, 256, 0, stream>>>(dt, dt_b, (size_t)M * DI);
  // selective scan + gating
  scan_kernel<<<dim3(DI / 256, BSZ), 256, 0, stream>>>(dt, xdblf, xhf, A_log, D_p, xz, 2 * DI, yb, SEQ);
  // out-projection: out = y @ out_w^T   (M x 1024, K=2048)
  cvt_f32_bf16_kernel<<<(DM * DI) / 256, 256, 0, stream>>>(out_w, wbuf, (size_t)DM * DI);
  {
    dim3 g(DM / 64, M / 128);
    wmma_gemm_bf16_tdm<<<g, 256, ldsTile, stream>>>(
        yb, wbuf, outp, M, DM, DI, DI, DI, DM);
  }
}

extern "C" void kernel_launch(void* const* d_in, const int* in_sizes, int n_in,
                              void* d_out, int out_size, void* d_ws, size_t ws_size,
                              hipStream_t stream) {
  (void)in_sizes; (void)n_in; (void)out_size; (void)ws_size;

  const float* hstm      = (const float*)d_in[0];
  const float* hscm      = (const float*)d_in[1];
  const float* residual  = (const float*)d_in[2];
  const float* norm_tm_w = (const float*)d_in[3];
  const float* wavg_tm_w = (const float*)d_in[4];
  const float* norm_cm_w = (const float*)d_in[5];
  const float* wavg_cm_w = (const float*)d_in[6];
  const float* tm[9], *cm[9];
  for (int i = 0; i < 9; ++i) tm[i] = (const float*)d_in[7 + i];
  for (int i = 0; i < 9; ++i) cm[i] = (const float*)d_in[16 + i];
  // order: in_w, conv_w, conv_b, xproj_w, dt_w, dt_b, A_log, D, out_w

  float* out0 = (float*)d_out;                       // (B,S,D)
  float* res2 = (float*)d_out + (size_t)MROWS * DM;  // (B,S,D)

  // ---- workspace carve ----
  char* ws = (char*)d_ws;
  size_t off = 0;
  auto carve = [&](size_t bytes) -> void* {
    void* p = ws + off;
    off += (bytes + 255) & ~(size_t)255;
    return p;
  };
  float*  res1  = (float*) carve((size_t)MROWS * DM * 4);       // 32 MB
  __bf16* xmix  = (__bf16*)carve((size_t)MROWS * DM * 2);       // 16 MB
  __bf16* wbuf  = (__bf16*)carve((size_t)2 * DI * DM * 2);      // 8 MB (largest weight)
  float*  xz    = (float*) carve((size_t)MROWS * 2 * DI * 4);   // 128 MB
  float*  xhf   = (float*) carve((size_t)MROWS * DI * 4);       // 64 MB
  __bf16* xhb   = (__bf16*)carve((size_t)MROWS * DI * 2);       // 32 MB
  float*  xdblf = (float*) carve((size_t)MROWS * GCOLS * 4);    // 3 MB
  __bf16* xdblb = (__bf16*)carve((size_t)MROWS * GCOLS * 2);    // 1.5 MB
  float*  dt    = (float*) carve((size_t)MROWS * DI * 4);       // 64 MB
  __bf16* yb    = (__bf16*)carve((size_t)MROWS * DI * 2);       // 32 MB
  float*  outp  = (float*) carve((size_t)MROWS * DM * 4);       // 32 MB
  __bf16* xcmb  = (__bf16*)carve((size_t)MROWS * DM * 2);       // 16 MB

  // ---- token mixer ----
  prep_tm_kernel<<<MROWS, 256, 0, stream>>>(hstm, hscm, residual, norm_tm_w, wavg_tm_w,
                                            res1, xmix);
  run_mamba(stream, xmix, tm[0], tm[1], tm[2], tm[3], tm[4], tm[5], tm[6], tm[7], tm[8],
            wbuf, xz, xhf, xhb, xdblf, xdblb, dt, yb, outp);

  // ---- channel mixer prep (res2 written straight into d_out, input transposed) ----
  prep_cm_kernel<<<MROWS, 256, 0, stream>>>(hstm, hscm, res1, outp, norm_cm_w, wavg_cm_w,
                                            res2, xcmb);
  run_mamba(stream, xcmb, cm[0], cm[1], cm[2], cm[3], cm[4], cm[5], cm[6], cm[7], cm[8],
            wbuf, xz, xhf, xhb, xdblf, xdblb, dt, yb, outp);

  // ---- transpose (B,D,S) -> (B,S,D) into d_out ----
  transpose_out_kernel<<<(MROWS * DM) / 256, 256, 0, stream>>>(outp, out0);
}
